// UCDM_25555055412008
// MI455X (gfx1250) — compile-verified, matrix operand order checked by reading, not run on hardware
//
#include <hip/hip_runtime.h>
#include <hip/hip_bf16.h>
#include <math.h>

// ---------------------------------------------------------------------------
// MI455X / gfx1250, wave32. f16 WMMA (v_wmma_f32_16x16x32_f16) for both GEMMs,
// f32 everywhere else. One 128-thread workgroup (4 waves) per batch element.
// ---------------------------------------------------------------------------

typedef __attribute__((ext_vector_type(16))) _Float16 v16h;
typedef __attribute__((ext_vector_type(8)))  _Float16 v8h;
typedef __attribute__((ext_vector_type(8)))  float    v8f;

#define HPAD 72   // f16 LDS row stride in halves (144B: 16B-aligned, conflict-free)
#define FPAD 68   // f32 LDS row stride in dwords (272B: 16B-aligned)

// Load a 16x32 f16 fragment (A, or B==Y^T as row-reads of Y) from LDS.
// ISA layout (CDNA5 ISA): lane&15 = matrix row; lanes>=16 take K+8; halves
// 0..7 = K[k0..k0+7], halves 8..15 = K[k0+16..k0+23].
__device__ __forceinline__ v16h ldsFrag(const _Float16* base, int rowTile,
                                        int kb, int lane) {
    const int r  = rowTile * 16 + (lane & 15);
    const int k0 = kb + ((lane >> 4) << 3);
    const _Float16* p = base + r * HPAD + k0;
    union { v16h v; v8h h[2]; } u;
    u.h[0] = *(const v8h*)(p);        // 16B -> ds_load_b128
    u.h[1] = *(const v8h*)(p + 16);
    return u.v;
}

// One attention block: out[d] = (1/L) * sum_m colsum(S)[m] * x[m][d]
__device__ __forceinline__ void att_block_dev(
    const int* __restrict__ seq,          // [L] indices for this batch element
    const float* __restrict__ table,      // [N,64] embedding table
    const float* __restrict__ pos,        // [L,64]
    const float* __restrict__ W,          // [64,64]
    const float* __restrict__ bias,       // [64]
    int L,
    _Float16* xh, _Float16* wh, _Float16* qh,   // LDS, [64*HPAD]
    float* xf,                                   // LDS, [64*FPAD]
    float* cpart,                                // LDS, [4*64]
    int*   sidx,                                 // LDS, [64]
    float* outv)                                 // LDS, [64]
{
    const int tid  = threadIdx.x;
    const int lane = tid & 31;
    const int wid  = tid >> 5;

    if (tid < 64) sidx[tid] = (tid < L) ? seq[tid] : 0;
    __syncthreads();

    // ---- gather x = table[seq]+pos (f32 + f16 copies), stage W as f16 ----
    #pragma unroll
    for (int it = 0; it < 8; ++it) {
        const int chunk = tid + it * 128;       // 1024 float4 chunks = 64x64
        const int r = chunk >> 4;
        const int c = (chunk & 15) << 2;
        float4 v = make_float4(0.f, 0.f, 0.f, 0.f);
        if (r < L) {
            const float4 e = *(const float4*)(table + (size_t)sidx[r] * 64 + c);
            const float4 p = *(const float4*)(pos + r * 64 + c);
            v = make_float4(e.x + p.x, e.y + p.y, e.z + p.z, e.w + p.w);
        }
        float* xr = xf + r * FPAD + c;
        xr[0] = v.x; xr[1] = v.y; xr[2] = v.z; xr[3] = v.w;
        _Float16* xr16 = xh + r * HPAD + c;
        xr16[0] = (_Float16)v.x; xr16[1] = (_Float16)v.y;
        xr16[2] = (_Float16)v.z; xr16[3] = (_Float16)v.w;

        const float4 w4 = *(const float4*)(W + (size_t)chunk * 4);
        _Float16* wr = wh + r * HPAD + c;
        wr[0] = (_Float16)w4.x; wr[1] = (_Float16)w4.y;
        wr[2] = (_Float16)w4.z; wr[3] = (_Float16)w4.w;
    }
    __syncthreads();

    // ---- GEMM1: q = relu(x @ W^T + b)  (64x64x64, wave wid owns M-tile) ----
    #pragma unroll
    for (int nt = 0; nt < 4; ++nt) {
        v8f acc = {};
        #pragma unroll
        for (int kb = 0; kb < 64; kb += 32) {
            v16h a  = ldsFrag(xh, wid, kb, lane);
            v16h bm = ldsFrag(wh, nt,  kb, lane);   // B = W^T via row-reads of W
            acc = __builtin_amdgcn_wmma_f32_16x16x32_f16(
                false, a, false, bm, (short)0, acc, false, false);
        }
        const int dcol  = nt * 16 + (lane & 15);
        const float bv  = bias[dcol];
        const int rbase = wid * 16 + ((lane >> 4) << 3);
        #pragma unroll
        for (int i = 0; i < 8; ++i) {
            float qv = acc[i] + bv;
            qv = qv > 0.f ? qv : 0.f;
            qh[(rbase + i) * HPAD + dcol] = (_Float16)qv;
        }
    }
    __syncthreads();

    // ---- GEMM2: aff = q @ q^T / 8; colsum(sigmoid with 0-diag, l<L mask) ----
    #pragma unroll
    for (int nt = 0; nt < 4; ++nt) {
        v8f acc = {};
        #pragma unroll
        for (int kb = 0; kb < 64; kb += 32) {
            v16h a  = ldsFrag(qh, wid, kb, lane);
            v16h bm = ldsFrag(qh, nt,  kb, lane);   // B = q^T via row-reads of q
            acc = __builtin_amdgcn_wmma_f32_16x16x32_f16(
                false, a, false, bm, (short)0, acc, false, false);
        }
        const int m = nt * 16 + (lane & 15);
        float csum = 0.f;
        #pragma unroll
        for (int i = 0; i < 8; ++i) {
            const int l = wid * 16 + i + ((lane >> 4) << 3);
            float s = 0.f;
            if (l < L)
                s = (l == m) ? 0.5f : 1.f / (1.f + __expf(-acc[i] * 0.125f));
            csum += s;
        }
        csum += __shfl_xor(csum, 16, 32);           // fold hi half-lanes
        if (lane < 16) cpart[wid * 64 + m] = csum;  // per-wave slice, no atomics
    }
    __syncthreads();

    // ---- combine: out[d] = (1/L) * sum_m c[m] * x[m][d] ----
    if (tid < 64) {
        float acc = 0.f;
        #pragma unroll 4
        for (int m = 0; m < 64; ++m) {
            const float c = cpart[m] + cpart[64 + m] + cpart[128 + m] + cpart[192 + m];
            acc = fmaf(c, xf[m * FPAD + tid], acc);
        }
        outv[tid] = acc * (1.f / (float)L);
    }
    __syncthreads();
}

__global__ __launch_bounds__(128)
void UCDM_25555055412008_kernel(
    const int*   __restrict__ seq_item,      // [B,L]
    const int*   __restrict__ cluster_seq,   // [B,L]
    const int*   __restrict__ target,        // [B,T]
    const int*   __restrict__ user_id,       // [B]
    const float* __restrict__ item_embed,    // [N,64]
    const float* __restrict__ item_embed_2,  // [N,64]
    const float* __restrict__ item_short,    // [N,64]
    const float* __restrict__ item_long,     // [N,64]
    const float* __restrict__ user_embed,    // [U,64]
    const float* __restrict__ W1, const float* __restrict__ b1,
    const float* __restrict__ W2, const float* __restrict__ b2,
    const float* __restrict__ pos_embed,     // [L,64]
    float* __restrict__ out,                 // [B,T]
    int L, int T)
{
    __shared__ __align__(16) _Float16 sh_xh[64 * HPAD];
    __shared__ __align__(16) _Float16 sh_wh[64 * HPAD];
    __shared__ __align__(16) _Float16 sh_qh[64 * HPAD];
    __shared__ __align__(16) float    sh_xf[64 * FPAD];
    __shared__ float sh_cpart[4 * 64];
    __shared__ int   sh_idx[64];
    __shared__ float sh_att[64];
    __shared__ float sh_cl[64];
    __shared__ float sh_u[64];

    const int b    = blockIdx.x;
    const int tid  = threadIdx.x;
    const int lane = tid & 31;
    const int wid  = tid >> 5;

    att_block_dev(seq_item + (size_t)b * L, item_embed, pos_embed, W1, b1, L,
                  sh_xh, sh_wh, sh_qh, sh_xf, sh_cpart, sh_idx, sh_att);
    att_block_dev(cluster_seq + (size_t)b * L, item_embed_2, pos_embed, W2, b2, L,
                  sh_xh, sh_wh, sh_qh, sh_xf, sh_cpart, sh_idx, sh_cl);

    if (tid < 64)
        sh_u[tid] = user_embed[(size_t)user_id[b] * 64 + tid];
    __syncthreads();

    // ---- distances: one wave per target, 2 dims per lane, wave32 reduce ----
    const int* targ = target + (size_t)b * T;
    float* yrow = out + (size_t)b * T;
    for (int t = wid; t < T; t += 4) {
        const int tgt = targ[t];
        const float* ts = item_short + (size_t)tgt * 64;
        const float* tl = item_long  + (size_t)tgt * 64;

        if (t + 4 < T) {                         // prefetch next target's rows
            const int tn = targ[t + 4];
            __builtin_prefetch(item_short + (size_t)tn * 64, 0, 0);
            __builtin_prefetch(item_long  + (size_t)tn * 64, 0, 0);
        }

        const float s0 = ts[lane], s1 = ts[lane + 32];
        const float l0 = tl[lane], l1 = tl[lane + 32];
        const float a0 = sh_att[lane] - s0, a1 = sh_att[lane + 32] - s1;
        const float c0 = sh_cl[lane]  - s0, c1 = sh_cl[lane + 32]  - s1;
        const float u0 = sh_u[lane]   - l0, u1 = sh_u[lane + 32]   - l1;
        float da = a0 * a0 + a1 * a1;
        float dc = c0 * c0 + c1 * c1;
        float dl = u0 * u0 + u1 * u1;
        #pragma unroll
        for (int off = 16; off > 0; off >>= 1) {
            da += __shfl_xor(da, off, 32);
            dc += __shfl_xor(dc, off, 32);
            dl += __shfl_xor(dl, off, 32);
        }
        if (lane == 0)
            yrow[t] = 0.1f * sqrtf(dl) + 0.8f * sqrtf(da) + 0.1f * sqrtf(dc);
    }
}

extern "C" void kernel_launch(void* const* d_in, const int* in_sizes, int n_in,
                              void* d_out, int out_size, void* d_ws, size_t ws_size,
                              hipStream_t stream) {
    (void)n_in; (void)d_ws; (void)ws_size; (void)out_size;
    const int* seq_item     = (const int*)  d_in[0];
    const int* cluster_seq  = (const int*)  d_in[1];
    const int* target       = (const int*)  d_in[2];
    const int* user_id      = (const int*)  d_in[3];
    const float* item_embed   = (const float*)d_in[4];
    const float* item_embed_2 = (const float*)d_in[5];
    const float* item_short   = (const float*)d_in[6];
    const float* item_long    = (const float*)d_in[7];
    const float* user_embed   = (const float*)d_in[8];
    const float* W1 = (const float*)d_in[9];
    const float* b1 = (const float*)d_in[10];
    const float* W2 = (const float*)d_in[11];
    const float* b2 = (const float*)d_in[12];
    const float* pos_embed = (const float*)d_in[13];

    const int B = in_sizes[3];          // user_id is [B]
    const int L = in_sizes[0] / B;      // 50
    const int T = in_sizes[2] / B;      // 100

    UCDM_25555055412008_kernel<<<dim3(B), dim3(128), 0, stream>>>(
        seq_item, cluster_seq, target, user_id,
        item_embed, item_embed_2, item_short, item_long, user_embed,
        W1, b1, W2, b2, pos_embed,
        (float*)d_out, L, T);
}
